// Transformer_58239756533890
// MI455X (gfx1250) — matrix operand order repaired and checked by reading, not compiled, not run on hardware
//
#include <hip/hip_runtime.h>
#include <hip/hip_bf16.h>
#include <stdint.h>

typedef _Float16 h2  __attribute__((ext_vector_type(2)));
typedef _Float16 h8  __attribute__((ext_vector_type(8)));
typedef _Float16 h16 __attribute__((ext_vector_type(16)));
typedef float    f8  __attribute__((ext_vector_type(8)));
typedef int      v4i __attribute__((ext_vector_type(4)));

#define GAS __attribute__((address_space(1)))
#define LAS __attribute__((address_space(3)))

#define NB      64          // batch
#define NTOK    196         // 14*14 tokens
#define CIN     384
#define COUT    768
#define HID     1536
#define INNER   256
#define NHEAD   8
#define HDIM    32
#define ROWS    (NB*NTOK)   // 12544
#define SCALE_F 0.17677669529663689f   // 32^-0.5
#define NPAD    224         // token dim padded to 7*32
#define MPAD    208         // row tiles: 13*16

// CDNA5 async global->LDS copy path (ASYNCcnt-tracked), probe-confirmed to
// exist on this toolchain (param0: AS1 int4*, param1: AS3 int4*).
#if defined(__has_builtin)
#if __has_builtin(__builtin_amdgcn_global_load_async_to_lds_b128) && \
    __has_builtin(__builtin_amdgcn_s_wait_asynccnt)
#define USE_ASYNC_COPY 1
#endif
#endif

// ---------------- WMMA helpers (CDNA5 16x16x32 f16 -> f32) ----------------
static __device__ inline f8 wmma_f16(h16 a, h16 b, f8 c) {
  return __builtin_amdgcn_wmma_f32_16x16x32_f16(false, a, false, b, (short)0, c,
                                                false, false);
}

// A fragment: row-major [M][ld] halves.  lane m=lane&15, half=lane>>4.
// element e -> K = (e/8)*16 + half*8 + e%8  (ISA 7.12.2, 16-bit A 16x32)
static __device__ inline h16 frag_a_h(const _Float16* base, int row, int ld) {
  int lane = threadIdx.x & 31;
  int m = lane & 15, hf = lane >> 4;
  const _Float16* p = base + (size_t)(row + m) * ld + hf * 8;
  h8 lo = *(const h8*)(p);
  h8 hi = *(const h8*)(p + 16);
  h16 r;
#pragma unroll
  for (int e = 0; e < 8; ++e) { r[e] = lo[e]; r[e + 8] = hi[e]; }
  return r;
}

// B fragment from N-major staging (Bt[n][k], ld halves per n-row).
// element e -> K = (e/8)*16 + half*8 + e%8
static __device__ inline h16 frag_b_h(const _Float16* base, int col, int ld) {
  int lane = threadIdx.x & 31;
  int n = lane & 15, hf = lane >> 4;
  const _Float16* p = base + (size_t)(col + n) * ld + hf * 8;
  h8 lo = *(const h8*)(p);
  h8 hi = *(const h8*)(p + 16);
  h16 r;
#pragma unroll
  for (int e = 0; e < 8; ++e) { r[e] = lo[e]; r[e + 8] = hi[e]; }
  return r;
}

// A fragment from fp32 source (softmax probabilities), convert to f16
static __device__ inline h16 frag_a_f32(const float* base, int row, int ld) {
  int lane = threadIdx.x & 31;
  int m = lane & 15, hf = lane >> 4;
  const float* p = base + (size_t)(row + m) * ld + hf * 8;
  h16 r;
#pragma unroll
  for (int e = 0; e < 8; ++e) {
    r[e]     = (_Float16)p[e];
    r[e + 8] = (_Float16)p[16 + e];
  }
  return r;
}

// 16-byte global->LDS stage (async path bypasses VGPRs entirely)
static __device__ inline void stage16(_Float16* dst_lds, const _Float16* src) {
#ifdef USE_ASYNC_COPY
  __builtin_amdgcn_global_load_async_to_lds_b128(
      (GAS v4i*)src, (LAS v4i*)dst_lds, 0, 0);
#else
  *(h8*)dst_lds = *(const h8*)src;
#endif
}
static __device__ inline void stage_fence() {
#ifdef USE_ASYNC_COPY
  __builtin_amdgcn_s_wait_asynccnt(0);
#endif
}

// ---------------- tiled WMMA GEMM, 128x64 block tile, 32x32 per wave -------
// B matrix is pre-transposed to N-major [N][K].
// EPI: 0 = store f16   1 = store f32   2 = f32 accum += D + bias
//      3 = f16 gelu(D + bias)          4 = final: out[b][c][n] = Cin + D + bias
#define BM 128
#define BN 64
#define BK 32
#define LDA 40   // padded half-stride (80 B, keeps 16B alignment)

template <int EPI>
__global__ __launch_bounds__(256) void gemm_epi(
    const _Float16* __restrict__ A, const _Float16* __restrict__ Bn,
    const float* __restrict__ bias, const float* __restrict__ Cin,
    void* __restrict__ Out, int M, int N, int K) {
  __shared__ __align__(16) _Float16 As[BM * LDA];
  __shared__ __align__(16) _Float16 Bs[BN * LDA];
  int tid = threadIdx.x;
  int wave = tid >> 5;
  int rowBase = blockIdx.y * BM;
  int colBase = blockIdx.x * BN;
  int mt = (wave >> 1) * 32;   // wave's 32-row slab within block tile
  int nt = (wave & 1) * 32;    // wave's 32-col slab

  // staging coords (16B segments: 4 per 64B row-slice)
  int ar0 = tid >> 2, asg = (tid & 3) * 8;      // A seg 0 (rows 0..63)
  int ar1 = ar0 + 64;                           // A seg 1 (rows 64..127)
  int br  = tid >> 2, bsg = asg;                // B seg (64 rows * 4 segs)

  f8 acc00 = {}, acc01 = {}, acc10 = {}, acc11 = {};
  for (int kk = 0; kk < K; kk += BK) {
    __syncthreads();
    stage16(As + ar0 * LDA + asg, A + (size_t)(rowBase + ar0) * K + kk + asg);
    stage16(As + ar1 * LDA + asg, A + (size_t)(rowBase + ar1) * K + kk + asg);
    stage16(Bs + br * LDA + bsg, Bn + (size_t)(colBase + br) * K + kk + bsg);
    if (kk + BK < K) {  // prefetch next K-slice (global_prefetch_b8)
      __builtin_prefetch(A + (size_t)(rowBase + ar0) * K + kk + BK + asg, 0, 1);
      __builtin_prefetch(Bn + (size_t)(colBase + br) * K + kk + BK + bsg, 0, 1);
    }
    stage_fence();
    __syncthreads();

    h16 a0 = frag_a_h(As, mt, LDA);
    h16 a1 = frag_a_h(As, mt + 16, LDA);
    h16 b0 = frag_b_h(Bs, nt, LDA);
    h16 b1 = frag_b_h(Bs, nt + 16, LDA);
    acc00 = wmma_f16(a0, b0, acc00);
    acc01 = wmma_f16(a0, b1, acc01);
    acc10 = wmma_f16(a1, b0, acc10);
    acc11 = wmma_f16(a1, b1, acc11);
  }

  // epilogue: element r of v8f -> M = r + 8*half, N = lane&15
  int lane = tid & 31, n = lane & 15, hf = lane >> 4;
#pragma unroll
  for (int sm = 0; sm < 2; ++sm) {
#pragma unroll
    for (int sn = 0; sn < 2; ++sn) {
      f8 acc = sm ? (sn ? acc11 : acc10) : (sn ? acc01 : acc00);
#pragma unroll
      for (int r = 0; r < 8; ++r) {
        int gr = rowBase + mt + sm * 16 + r + 8 * hf;
        int gc = colBase + nt + sn * 16 + n;
        float v = acc[r];
        if (EPI == 0) {
          ((_Float16*)Out)[(size_t)gr * N + gc] = (_Float16)v;
        } else if (EPI == 1) {
          ((float*)Out)[(size_t)gr * N + gc] = v;
        } else if (EPI == 2) {
          float* o = (float*)Out;
          size_t i = (size_t)gr * N + gc;
          o[i] = o[i] + v + bias[gc];
        } else if (EPI == 3) {
          float g = v + bias[gc];
          g = 0.5f * g * (1.0f + erff(g * 0.70710678118654752f));
          ((_Float16*)Out)[(size_t)gr * N + gc] = (_Float16)g;
        } else if (EPI == 4) {
          int bb = gr / NTOK, nn = gr - bb * NTOK;
          ((float*)Out)[((size_t)bb * COUT + gc) * NTOK + nn] =
              Cin[(size_t)gr * COUT + gc] + v + bias[gc];
        }
      }
    }
  }
}

// ---------------- attention: one workgroup per (batch, head) ----------------
#define ATT_SMEM (MPAD * NPAD * 4 + (MPAD * HDIM + NPAD * HDIM + HDIM * NPAD) * 2)

__global__ __launch_bounds__(256) void attn_kernel(
    const _Float16* __restrict__ qkv,     // [ROWS][768] = q|k|v
    const float* __restrict__ biasHIJ,    // [8][196][196]
    _Float16* __restrict__ Oout) {        // [ROWS][256]
  extern __shared__ char smem[];
  float*    S  = (float*)smem;                          // [208][224] fp32
  _Float16* Qs = (_Float16*)(smem + MPAD * NPAD * 4);   // [208][32]
  _Float16* Ks = Qs + MPAD * HDIM;                      // [224][32]
  _Float16* Vt = Ks + NPAD * HDIM;                      // [32][224]
  int b = blockIdx.x >> 3;
  int h = blockIdx.x & 7;
  int tid = threadIdx.x, wave = tid >> 5;

  const _Float16* qb = qkv + (size_t)b * NTOK * (3 * INNER) + h * HDIM;
  for (int idx = tid; idx < NPAD * HDIM; idx += 256) {
    int j = idx >> 5, d = idx & 31;
    _Float16 kv = (j < NTOK) ? qb[(size_t)j * 768 + INNER + d]     : (_Float16)0.f;
    _Float16 vv = (j < NTOK) ? qb[(size_t)j * 768 + 2 * INNER + d] : (_Float16)0.f;
    Ks[idx] = kv;
    Vt[d * NPAD + j] = vv;
    if (j < MPAD)
      Qs[idx] = (j < NTOK) ? qb[(size_t)j * 768 + d] : (_Float16)0.f;
  }
  __syncthreads();

  // S = Q @ K^T  (raw dots, 13x14 tiles, K=32 in one WMMA)
  for (int t = wave; t < 13 * 14; t += 8) {
    int tm = (t / 14) * 16, tn = (t % 14) * 16;
    f8 acc = {};
    h16 a  = frag_a_h(Qs, tm, HDIM);
    h16 bf = frag_b_h(Ks, tn, HDIM);   // Ks row-major [j][d] == N-major for S
    acc = wmma_f16(a, bf, acc);
    int lane = tid & 31, n = lane & 15, hf = lane >> 4;
#pragma unroll
    for (int r = 0; r < 8; ++r)
      S[(size_t)(tm + r + 8 * hf) * NPAD + tn + n] = acc[r];
  }
  __syncthreads();

  // fp32 softmax with scale + relative-position bias (rows < 196)
  for (int r = tid; r < NTOK; r += 256) {
    float* row = S + (size_t)r * NPAD;
    const float* br = biasHIJ + ((size_t)h * NTOK + r) * NTOK;
    float mx = -1e30f;
    for (int c = 0; c < NTOK; ++c) {
      float t2 = row[c] * SCALE_F + br[c];
      row[c] = t2;
      mx = fmaxf(mx, t2);
    }
    float sum = 0.f;
    for (int c = 0; c < NTOK; ++c) {
      float e = expf(row[c] - mx);
      row[c] = e;
      sum += e;
    }
    float inv = 1.0f / sum;
    for (int c = 0; c < NTOK; ++c) row[c] *= inv;
    for (int c = NTOK; c < NPAD; ++c) row[c] = 0.f;
  }
  __syncthreads();

  // O = P @ V   (13x2 tiles, K=224 in 7 WMMA steps)
  for (int t = wave; t < 13 * 2; t += 8) {
    int tm = (t >> 1) * 16, tn = (t & 1) * 16;
    f8 acc = {};
#pragma unroll
    for (int kk = 0; kk < NPAD; kk += 32) {
      h16 a  = frag_a_f32(S + kk, tm, NPAD);
      h16 bf = frag_b_h(Vt + kk, tn, NPAD);   // Vt[d][j] is N-major
      acc = wmma_f16(a, bf, acc);
    }
    int lane = tid & 31, n = lane & 15, hf = lane >> 4;
#pragma unroll
    for (int r = 0; r < 8; ++r) {
      int row = tm + r + 8 * hf;
      if (row < NTOK)
        Oout[((size_t)b * NTOK + row) * INNER + h * HDIM + tn + n] =
            (_Float16)acc[r];
    }
  }
}

// ---------------- small prep / pointwise kernels ----------------
__global__ void f32_to_f16_k(const float* __restrict__ in,
                             _Float16* __restrict__ out, int n) {
  int i = blockIdx.x * 256 + threadIdx.x;
  if (i < n) out[i] = (_Float16)in[i];
}

// in [R][C] row-major f32 -> out [C][R] f16  (weights to N-major)
__global__ void transpose_f16_k(const float* __restrict__ in,
                                _Float16* __restrict__ out, int R, int C) {
  int i = blockIdx.x * 256 + threadIdx.x;
  if (i >= R * C) return;
  int r = i / C, c = i - r * C;
  out[(size_t)c * R + r] = (_Float16)in[i];
}

// bias[h][i][j] = table[((yi-yj+13)*27 + (xi-xj+13))*8 + h]
__global__ void build_bias_k(const float* __restrict__ table,
                             float* __restrict__ out) {
  int i = blockIdx.x * 256 + threadIdx.x;
  if (i >= NHEAD * NTOK * NTOK) return;
  int h = i / (NTOK * NTOK);
  int rem = i - h * NTOK * NTOK;
  int ti = rem / NTOK, tj = rem - ti * NTOK;
  int dy = ti / 14 - tj / 14 + 13;
  int dx = ti % 14 - tj % 14 + 13;
  out[i] = table[(dy * 27 + dx) * NHEAD + h];
}

// maxpool 3x3 s2 p1 : x[64][384][28][28] -> traw[b*196+n][384]
__global__ void pool_k(const float* __restrict__ x, float* __restrict__ traw) {
  int idx = blockIdx.x * 256 + threadIdx.x;
  if (idx >= NB * CIN * NTOK) return;
  int n = idx % NTOK;
  int c = (idx / NTOK) % CIN;
  int b = idx / (NTOK * CIN);
  int i = n / 14, j = n % 14;
  const float* xp = x + ((size_t)b * CIN + c) * 784;
  float m = -3.4e38f;
#pragma unroll
  for (int di = 0; di < 3; ++di) {
    int y = 2 * i + di - 1;
    if (y < 0 || y > 27) continue;
#pragma unroll
    for (int dj = 0; dj < 3; ++dj) {
      int xj = 2 * j + dj - 1;
      if (xj < 0 || xj > 27) continue;
      m = fmaxf(m, xp[y * 28 + xj]);
    }
  }
  traw[((size_t)b * NTOK + n) * CIN + c] = m;
}

// row LayerNorm; optionally also emit raw f16 copy
__global__ __launch_bounds__(128) void ln_rows_k(
    const float* __restrict__ in, const float* __restrict__ gam,
    const float* __restrict__ bet, _Float16* __restrict__ outln,
    _Float16* __restrict__ outraw, int C) {
  __shared__ float red[128];
  int row = blockIdx.x, tid = threadIdx.x;
  const float* rp = in + (size_t)row * C;
  float s = 0.f;
  for (int c = tid; c < C; c += 128) s += rp[c];
  red[tid] = s; __syncthreads();
  for (int o = 64; o > 0; o >>= 1) {
    if (tid < o) red[tid] += red[tid + o];
    __syncthreads();
  }
  float mu = red[0] / C;
  __syncthreads();
  float v = 0.f;
  for (int c = tid; c < C; c += 128) { float d = rp[c] - mu; v += d * d; }
  red[tid] = v; __syncthreads();
  for (int o = 64; o > 0; o >>= 1) {
    if (tid < o) red[tid] += red[tid + o];
    __syncthreads();
  }
  float inv = rsqrtf(red[0] / C + 1e-5f);
  for (int c = tid; c < C; c += 128) {
    float val = rp[c];
    outln[(size_t)row * C + c] = (_Float16)((val - mu) * inv * gam[c] + bet[c]);
    if (outraw) outraw[(size_t)row * C + c] = (_Float16)val;
  }
}

// ---------------- host ----------------
extern "C" void kernel_launch(void* const* d_in, const int* in_sizes, int n_in,
                              void* d_out, int out_size, void* d_ws,
                              size_t ws_size, hipStream_t stream) {
  (void)in_sizes; (void)n_in; (void)out_size; (void)ws_size;
  const float* x      = (const float*)d_in[0];
  const float* W_qkv  = (const float*)d_in[1];
  const float* bt     = (const float*)d_in[2];
  const float* W_out  = (const float*)d_in[3];
  const float* b_out  = (const float*)d_in[4];
  const float* ln1_g  = (const float*)d_in[5];
  const float* ln1_b  = (const float*)d_in[6];
  const float* ln2_g  = (const float*)d_in[7];
  const float* ln2_b  = (const float*)d_in[8];
  const float* W_fc1  = (const float*)d_in[9];
  const float* b_fc1  = (const float*)d_in[10];
  const float* W_fc2  = (const float*)d_in[11];
  const float* b_fc2  = (const float*)d_in[12];
  const float* W_proj = (const float*)d_in[13];
  float* out = (float*)d_out;

  char* ws = (char*)d_ws;
  size_t off = 0;
  auto take = [&](size_t bytes) {
    size_t o = off;
    off = (off + bytes + 255) & ~(size_t)255;
    return ws + o;
  };
  float*    traw32  = (float*)take((size_t)ROWS * CIN * 4);
  _Float16* traw16  = (_Float16*)take((size_t)ROWS * CIN * 2);
  _Float16* tln16   = (_Float16*)take((size_t)ROWS * CIN * 2);
  _Float16* qkv16   = (_Float16*)take((size_t)ROWS * 768 * 2);
  _Float16* attnO16 = (_Float16*)take((size_t)ROWS * INNER * 2);
  float*    h32     = (float*)take((size_t)ROWS * COUT * 4);
  _Float16* hln16   = (_Float16*)take((size_t)ROWS * COUT * 2);
  _Float16* u16     = (_Float16*)take((size_t)ROWS * HID * 2);
  _Float16* wqkvT   = (_Float16*)take((size_t)CIN * 768 * 2);    // [768][384]
  _Float16* wproj16 = (_Float16*)take((size_t)COUT * CIN * 2);   // [768][384]
  _Float16* woutT   = (_Float16*)take((size_t)INNER * COUT * 2); // [768][256]
  _Float16* wfc1T   = (_Float16*)take((size_t)COUT * HID * 2);   // [1536][768]
  _Float16* wfc2T   = (_Float16*)take((size_t)HID * COUT * 2);   // [768][1536]
  float*    biasHIJ = (float*)take((size_t)NHEAD * NTOK * NTOK * 4);

  auto grid1 = [](int n) { return dim3((n + 255) / 256); };

  // weight prep: all B matrices to N-major [N][K] f16
  transpose_f16_k<<<grid1(CIN * 768), 256, 0, stream>>>(W_qkv, wqkvT, CIN, 768);
  transpose_f16_k<<<grid1(INNER * COUT), 256, 0, stream>>>(W_out, woutT, INNER, COUT);
  transpose_f16_k<<<grid1(COUT * HID), 256, 0, stream>>>(W_fc1, wfc1T, COUT, HID);
  transpose_f16_k<<<grid1(HID * COUT), 256, 0, stream>>>(W_fc2, wfc2T, HID, COUT);
  f32_to_f16_k<<<grid1(COUT * CIN), 256, 0, stream>>>(W_proj, wproj16, COUT * CIN);
  build_bias_k<<<grid1(NHEAD * NTOK * NTOK), 256, 0, stream>>>(bt, biasHIJ);

  // pool + LN1
  pool_k<<<grid1(NB * CIN * NTOK), 256, 0, stream>>>(x, traw32);
  ln_rows_k<<<ROWS, 128, 0, stream>>>(traw32, ln1_g, ln1_b, tln16, traw16, CIN);

  // qkv GEMM  [12544,384] x [384,768] -> f16
  gemm_epi<0><<<dim3(768 / BN, ROWS / BM), 256, 0, stream>>>(
      tln16, wqkvT, nullptr, nullptr, qkv16, ROWS, 768, CIN);

  // attention (228 KB dynamic LDS per workgroup)
  (void)hipFuncSetAttribute((const void*)attn_kernel,
                            hipFuncAttributeMaxDynamicSharedMemorySize,
                            ATT_SMEM);
  attn_kernel<<<NB * NHEAD, 256, ATT_SMEM, stream>>>(qkv16, biasHIJ, attnO16);

  // proj 1x1 conv GEMM -> h (fp32)
  gemm_epi<1><<<dim3(COUT / BN, ROWS / BM), 256, 0, stream>>>(
      traw16, wproj16, nullptr, nullptr, h32, ROWS, COUT, CIN);
  // attention out-proj: h += O @ W_out + b_out
  gemm_epi<2><<<dim3(COUT / BN, ROWS / BM), 256, 0, stream>>>(
      attnO16, woutT, b_out, nullptr, h32, ROWS, COUT, INNER);

  // FFN
  ln_rows_k<<<ROWS, 128, 0, stream>>>(h32, ln2_g, ln2_b, hln16, nullptr, COUT);
  gemm_epi<3><<<dim3(HID / BN, ROWS / BM), 256, 0, stream>>>(
      hln16, wfc1T, b_fc1, nullptr, u16, ROWS, HID, COUT);
  gemm_epi<4><<<dim3(COUT / BN, ROWS / BM), 256, 0, stream>>>(
      u16, wfc2T, b_fc2, h32, out, ROWS, COUT, HID);
}
